// reader1_44796508897310
// MI455X (gfx1250) — matrix operand order repaired and checked by reading, not compiled, not run on hardware
//
#include <hip/hip_runtime.h>
#include <hip/hip_bf16.h>
#include <math.h>

// ---------------------------------------------------------------------------
// Problem constants (from reference): B=4, S=512, N=64, D=768, H=8, DK=64, L=2
// ---------------------------------------------------------------------------
#define BB 4
#define SS 512
#define NN 64
#define DD 768
#define HH 8
#define DKK 64
#define HDK 512        // H*DK
#define DINNER_ 768
#define LL 2

typedef __attribute__((ext_vector_type(16))) __bf16 v16bf;
typedef __attribute__((ext_vector_type(8)))  float  v8f;

static __device__ __forceinline__ __bf16 f2bf(float x) { return (__bf16)x; }

// ---------------------------------------------------------------------------
// Block reduction (sum) helper; blockDim.x must be 256 (8 waves of 32).
// ---------------------------------------------------------------------------
static __device__ __forceinline__ float block_reduce_sum(float v, float* sred) {
    __syncthreads();                       // protect sred reuse across calls
    int lane = threadIdx.x & 31;
    int w    = threadIdx.x >> 5;
    #pragma unroll
    for (int off = 16; off > 0; off >>= 1) v += __shfl_down(v, off, 32);
    if (lane == 0) sred[w] = v;
    __syncthreads();
    float r = (threadIdx.x < 8) ? sred[threadIdx.x] : 0.0f;
    if (w == 0) {
        #pragma unroll
        for (int off = 4; off > 0; off >>= 1) r += __shfl_down(r, off, 32);
        if (lane == 0) sred[0] = r;
    }
    __syncthreads();
    return sred[0];
}

// ---------------------------------------------------------------------------
// c_flw[b,n,d] = max_s ( map[b,s,n] ? qc[b,s,d] : 0 )
// One block per (b,n); mapping column staged in LDS. qc is L2-resident.
// ---------------------------------------------------------------------------
__global__ __launch_bounds__(256) void cflw_kernel(const float* __restrict__ qc,
                                                   const int*   __restrict__ amap,
                                                   float*       __restrict__ out) {
    int b = blockIdx.x / NN;
    int n = blockIdx.x % NN;
    __shared__ int ms[SS];
    for (int s = threadIdx.x; s < SS; s += 256)
        ms[s] = amap[((size_t)b * SS + s) * NN + n];
    __syncthreads();
    const float* qp = qc + (size_t)b * SS * DD;
    for (int d = threadIdx.x; d < DD; d += 256) {
        float m = -INFINITY;
        for (int s = 0; s < SS; ++s) {
            float v = ms[s] ? qp[(size_t)s * DD + d] : 0.0f;
            m = fmaxf(m, v);
        }
        out[(size_t)blockIdx.x * DD + d] = m;
    }
}

// qmax[b,d] = max_s qc[b,s,d]
__global__ void qmax_kernel(const float* __restrict__ qc, float* __restrict__ out) {
    int idx = blockIdx.x * blockDim.x + threadIdx.x;
    if (idx >= BB * DD) return;
    int b = idx / DD, d = idx % DD;
    float m = -INFINITY;
    const float* qp = qc + (size_t)b * SS * DD + d;
    for (int s = 0; s < SS; ++s) m = fmaxf(m, qp[(size_t)s * DD]);
    out[idx] = m;
}

// out = c_flw + sinusoid PE (positions 0..NN-1)
__global__ void pe_add_kernel(const float* __restrict__ cflw, float* __restrict__ out) {
    int idx = blockIdx.x * blockDim.x + threadIdx.x;
    if (idx >= BB * NN * DD) return;
    int d = idx % DD;
    int n = (idx / DD) % NN;
    float di  = (float)(2 * (d / 2)) / (float)DD;
    float ang = (float)n * __powf(10000.0f, -di);
    float pe  = (d & 1) ? __cosf(ang) : __sinf(ang);
    out[idx] = cflw[idx] + pe;
}

// ---------------------------------------------------------------------------
// Row LayerNorm with optional residual: out = LN(x + res) * g + b.  D = 768,
// 256 threads -> 3 elements/thread, two-pass mean/variance.
// ---------------------------------------------------------------------------
__global__ __launch_bounds__(256) void ln_kernel(const float* __restrict__ x,
                                                 const float* __restrict__ res,
                                                 const float* __restrict__ g,
                                                 const float* __restrict__ bta,
                                                 float*       __restrict__ out) {
    __shared__ float sred[8];
    size_t row = blockIdx.x;
    float v[3];
    float s = 0.0f;
    #pragma unroll
    for (int j = 0; j < 3; ++j) {
        int idx = threadIdx.x + j * 256;
        float t = x[row * DD + idx];
        if (res) t += res[row * DD + idx];
        v[j] = t;
        s += t;
    }
    float mu = block_reduce_sum(s, sred) * (1.0f / DD);
    float s2 = 0.0f;
    #pragma unroll
    for (int j = 0; j < 3; ++j) { float dv = v[j] - mu; s2 += dv * dv; }
    float var = block_reduce_sum(s2, sred) * (1.0f / DD);
    float inv = rsqrtf(var + 1e-6f);
    #pragma unroll
    for (int j = 0; j < 3; ++j) {
        int idx = threadIdx.x + j * 256;
        out[row * DD + idx] = (v[j] - mu) * inv * g[idx] + bta[idx];
    }
}

// ---------------------------------------------------------------------------
// Generic GEMM  C[M,N] = A[M,K] @ W[K,N] (+bias, optional ReLU), fp32 in/out,
// bf16 WMMA compute.  64x64 C-tile per 256-thread block (8 waves in 4x2 grid,
// each wave owns a 16x32 strip -> two v8f accumulators).  K stepped by 32.
// B tile is stored TRANSPOSED in LDS so every WMMA operand fragment is two
// contiguous 16-byte runs -> ds_load_b128.  Next K-slab is prefetched
// (global_prefetch_b8) while the current one computes.
// ---------------------------------------------------------------------------
template <int RELU>
__global__ __launch_bounds__(256) void gemm_bf16_wmma(const float* __restrict__ A,
                                                      const float* __restrict__ W,
                                                      const float* __restrict__ bias,
                                                      float*       __restrict__ C,
                                                      int M, int N, int K) {
    __shared__ __bf16 As[64][40];    // [row][k]  row stride 80B (16B multiple)
    __shared__ __bf16 BsT[64][40];   // [col][k]  transposed B tile

    int tid  = threadIdx.x;
    int wave = tid >> 5;
    int lane = tid & 31;
    int wr   = wave >> 1;            // 0..3 : 16-row strip
    int wc   = wave & 1;             // 0..1 : 32-col strip
    int bm   = blockIdx.x * 64;
    int bn   = blockIdx.y * 64;

    v8f acc0 = {}; v8f acc1 = {};

    int m     = lane & 15;
    int koff  = (lane >> 4) * 16;    // A operand K half
    int khalf = (lane >> 4) * 16;    // B operand K half

    for (int k0 = 0; k0 < K; k0 += 32) {
        for (int i = tid; i < 64 * 32; i += 256) {
            int r = i >> 5, c = i & 31;
            As[r][c] = f2bf(A[(size_t)(bm + r) * K + k0 + c]);
        }
        for (int i = tid; i < 32 * 64; i += 256) {
            int r = i >> 6, c = i & 63;                    // coalesced global read
            BsT[c][r] = f2bf(W[(size_t)(k0 + r) * N + bn + c]);
        }
        if (k0 + 32 < K) {           // prefetch next K-slab into cache
            int r = tid >> 2;        // 64 rows, 4 threads/row
            __builtin_prefetch(&A[(size_t)(bm + r) * K + k0 + 32], 0, 1);
            int kr = tid >> 3;       // 32 rows, 8 threads/row
            __builtin_prefetch(&W[(size_t)(k0 + 32 + kr) * N + bn], 0, 1);
        }
        __syncthreads();

        v16bf a;
        #pragma unroll
        for (int i = 0; i < 8; ++i) {
            int kb = 2 * i;
            a[2 * i]     = As[wr * 16 + m][koff + kb];
            a[2 * i + 1] = As[wr * 16 + m][koff + kb + 1];
        }
        v16bf b0, b1;
        #pragma unroll
        for (int e = 0; e < 16; ++e) {
            b0[e] = BsT[wc * 32 + m][khalf + e];           // contiguous -> b128
            b1[e] = BsT[wc * 32 + 16 + m][khalf + e];
        }
        acc0 = __builtin_amdgcn_wmma_f32_16x16x32_bf16(false, a, false, b0, (short)0, acc0, false, false);
        acc1 = __builtin_amdgcn_wmma_f32_16x16x32_bf16(false, a, false, b1, (short)0, acc1, false, false);
        __syncthreads();
    }

    int lh = lane >> 4;
    #pragma unroll
    for (int r = 0; r < 8; ++r) {
        int row = bm + wr * 16 + r + 8 * lh;      // C/D layout: VGPR r -> M=r / r+8
        int c0  = bn + wc * 32 + m;
        int c1  = c0 + 16;
        float v0 = acc0[r] + (bias ? bias[c0] : 0.0f);
        float v1 = acc1[r] + (bias ? bias[c1] : 0.0f);
        if (RELU) { v0 = fmaxf(v0, 0.0f); v1 = fmaxf(v1, 0.0f); }
        C[(size_t)row * N + c0] = v0;
        C[(size_t)row * N + c1] = v1;
    }
}

// ---------------------------------------------------------------------------
// Fused attention for one (b,h): scores = Q K^T * 0.125, key-mask, softmax,
// O = P V.  Both matmuls via bf16 WMMA.  128 threads = 4 waves, each wave a
// 16-row strip with 4 accumulators.  V is stored transposed (VsT[dk][key]) so
// the P·V B-fragment is row-contiguous -> ds_load_b128.
// ---------------------------------------------------------------------------
__global__ __launch_bounds__(128) void attn_wmma(const float* __restrict__ Q,
                                                 const float* __restrict__ Kb,
                                                 const float* __restrict__ V,
                                                 const int*   __restrict__ smap,
                                                 float*       __restrict__ O,
                                                 float*       __restrict__ att_out,
                                                 int acc_att) {
    int b = blockIdx.x / HH;
    int h = blockIdx.x % HH;

    __shared__ __bf16 Qs[64][72];    // [query][k]
    __shared__ __bf16 Ks[64][72];    // [key][k]   (scores B-frag reads rows)
    __shared__ __bf16 VsT[64][72];   // [dk][key]  (PV B-frag reads rows)
    __shared__ float  Sc[64][66];
    __shared__ float  attsum[64];
    __shared__ int    msk[64];

    int tid = threadIdx.x;
    for (int i = tid; i < 64 * 64; i += 128) {
        int r = i >> 6, c = i & 63;
        size_t g = ((size_t)(b * NN + r)) * HDK + h * DKK + c;
        Qs[r][c]  = f2bf(Q[g]);
        Ks[r][c]  = f2bf(Kb[g]);
        VsT[c][r] = f2bf(V[g]);      // transpose on store
    }
    if (tid < 64) { msk[tid] = smap[b * NN + tid]; attsum[tid] = 0.0f; }
    __syncthreads();

    int wave  = tid >> 5;
    int lane  = tid & 31;
    int m     = lane & 15;
    int koff  = (lane >> 4) * 16;
    int khalf = (lane >> 4) * 16;
    int lh    = lane >> 4;

    // ---- scores = Q @ K^T ----
    v8f acc[4] = {{}, {}, {}, {}};
    #pragma unroll
    for (int k0 = 0; k0 < 64; k0 += 32) {
        v16bf a;
        #pragma unroll
        for (int i = 0; i < 8; ++i) {
            int kb = 2 * i;
            a[2 * i]     = Qs[wave * 16 + m][k0 + koff + kb];
            a[2 * i + 1] = Qs[wave * 16 + m][k0 + koff + kb + 1];
        }
        #pragma unroll
        for (int t = 0; t < 4; ++t) {
            v16bf bb;
            #pragma unroll
            for (int e = 0; e < 16; ++e)
                bb[e] = Ks[t * 16 + m][k0 + khalf + e];   // row-contiguous
            acc[t] = __builtin_amdgcn_wmma_f32_16x16x32_bf16(false, a, false, bb, (short)0, acc[t], false, false);
        }
    }
    #pragma unroll
    for (int t = 0; t < 4; ++t)
        #pragma unroll
        for (int r = 0; r < 8; ++r) {
            int row = wave * 16 + r + 8 * lh;
            int col = t * 16 + m;
            float s = acc[t][r] * 0.125f;                 // 1/sqrt(64)
            Sc[row][col] = msk[col] ? s : -1.0e9f;
        }
    __syncthreads();

    // ---- per-row softmax (+ att accumulation) ----
    if (tid < 64) {
        float mx = -INFINITY;
        for (int k = 0; k < 64; ++k) mx = fmaxf(mx, Sc[tid][k]);
        float sum = 0.0f;
        for (int k = 0; k < 64; ++k) { float e = __expf(Sc[tid][k] - mx); Sc[tid][k] = e; sum += e; }
        float inv = 1.0f / sum;
        for (int k = 0; k < 64; ++k) {
            float p = Sc[tid][k] * inv;
            Sc[tid][k] = p;
            if (acc_att) atomicAdd(&attsum[k], p);
        }
    }
    __syncthreads();

    // ---- O = P @ V ----
    v8f oacc[4] = {{}, {}, {}, {}};
    #pragma unroll
    for (int k0 = 0; k0 < 64; k0 += 32) {
        v16bf a;
        #pragma unroll
        for (int i = 0; i < 8; ++i) {
            int kb = 2 * i;
            a[2 * i]     = f2bf(Sc[wave * 16 + m][k0 + koff + kb]);
            a[2 * i + 1] = f2bf(Sc[wave * 16 + m][k0 + koff + kb + 1]);
        }
        #pragma unroll
        for (int t = 0; t < 4; ++t) {
            v16bf bb;
            #pragma unroll
            for (int e = 0; e < 16; ++e)
                bb[e] = VsT[t * 16 + m][k0 + khalf + e];  // row-contiguous
            oacc[t] = __builtin_amdgcn_wmma_f32_16x16x32_bf16(false, a, false, bb, (short)0, oacc[t], false, false);
        }
    }
    #pragma unroll
    for (int t = 0; t < 4; ++t)
        #pragma unroll
        for (int r = 0; r < 8; ++r) {
            int row = wave * 16 + r + 8 * lh;
            int col = t * 16 + m;
            O[((size_t)(b * NN + row)) * HDK + h * DKK + col] = oacc[t][r];
        }

    if (acc_att) {
        __syncthreads();
        if (tid < 64)
            atomicAdd(&att_out[b * NN + tid], attsum[tid] * (1.0f / (HH * 64.0f)));
    }
}

// sp_logits[row] = h[row,:]·sp_w + sp_b   (row = b*NN + n)
__global__ __launch_bounds__(256) void sp_kernel(const float* __restrict__ hf,
                                                 const float* __restrict__ spw,
                                                 const float* __restrict__ spb,
                                                 float*       __restrict__ out) {
    __shared__ float sred[8];
    size_t row = blockIdx.x;
    float s = 0.0f;
    for (int d = threadIdx.x; d < DD; d += 256) s += hf[row * DD + d] * spw[d];
    float tot = block_reduce_sum(s, sred);
    if (threadIdx.x == 0) out[row] = tot + spb[0];
}

// type_logits[b,t] = qmax[b,:]·type_w[t,:] + type_b[t]
__global__ __launch_bounds__(256) void type_kernel(const float* __restrict__ qm,
                                                   const float* __restrict__ tw,
                                                   const float* __restrict__ tb,
                                                   float*       __restrict__ out) {
    __shared__ float sred[8];
    int b = blockIdx.x / 3, t = blockIdx.x % 3;
    float s = 0.0f;
    for (int d = threadIdx.x; d < DD; d += 256) s += qm[b * DD + d] * tw[t * DD + d];
    float tot = block_reduce_sum(s, sred);
    if (threadIdx.x == 0) out[b * 3 + t] = tot + tb[t];
}

// ---------------------------------------------------------------------------
// Host orchestration (graph-capture safe: only async launches/copies/memset).
// Input order follows setup_inputs() insertion order with tparams flattened
// in its own insertion order.
// ---------------------------------------------------------------------------
extern "C" void kernel_launch(void* const* d_in, const int* in_sizes, int n_in,
                              void* d_out, int out_size, void* d_ws, size_t ws_size,
                              hipStream_t stream) {
    (void)in_sizes; (void)n_in; (void)out_size; (void)ws_size;

    const float* qc    = (const float*)d_in[0];
    const int*   amap  = (const int*)d_in[1];
    const int*   smap  = (const int*)d_in[2];
    const float* ln0_g = (const float*)d_in[3];
    const float* ln0_b = (const float*)d_in[4];
    const float* wq    = (const float*)d_in[5];
    const float* bq    = (const float*)d_in[6];
    const float* wk    = (const float*)d_in[7];
    const float* bk    = (const float*)d_in[8];
    const float* wv    = (const float*)d_in[9];
    const float* bv    = (const float*)d_in[10];
    const float* fcw   = (const float*)d_in[11];
    const float* fcb   = (const float*)d_in[12];
    const float* ln1_g = (const float*)d_in[13];
    const float* ln1_b = (const float*)d_in[14];
    const float* w1    = (const float*)d_in[15];
    const float* b1    = (const float*)d_in[16];
    const float* w2    = (const float*)d_in[17];
    const float* b2    = (const float*)d_in[18];
    const float* ln2_g = (const float*)d_in[19];
    const float* ln2_b = (const float*)d_in[20];
    const float* spw   = (const float*)d_in[21];
    const float* spb   = (const float*)d_in[22];
    const float* tw    = (const float*)d_in[23];
    const float* tb    = (const float*)d_in[24];

    const size_t QC_ELEMS = (size_t)BB * SS * DD;   // 1,572,864
    float* out_type = (float*)d_out;                // 12
    float* out_qc   = out_type + 12;
    float* out_sp   = out_qc + QC_ELEMS;            // 256
    float* out_att  = out_sp + BB * NN;             // 256

    const size_t ROWE = (size_t)BB * NN * DD;       // 196,608
    const size_t QKVE = (size_t)BB * NN * HDK;      // 131,072
    float* ws   = (float*)d_ws;
    float* cflw = ws;
    float* hbuf = cflw + ROWE;
    float* h2   = hbuf + ROWE;
    float* tmp  = h2   + ROWE;
    float* ff1  = tmp  + ROWE;
    float* qb   = ff1  + ROWE;
    float* kb   = qb   + QKVE;
    float* vb   = kb   + QKVE;
    float* ob   = vb   + QKVE;
    float* qm   = ob   + QKVE;                      // B*D

    // qc_out passthrough + att init
    hipMemcpyAsync(out_qc, qc, QC_ELEMS * sizeof(float), hipMemcpyDeviceToDevice, stream);
    hipMemsetAsync(out_att, 0, BB * NN * sizeof(float), stream);

    cflw_kernel<<<BB * NN, 256, 0, stream>>>(qc, amap, cflw);
    qmax_kernel<<<(BB * DD + 255) / 256, 256, 0, stream>>>(qc, qm);
    pe_add_kernel<<<(int)((ROWE + 255) / 256), 256, 0, stream>>>(cflw, tmp);
    ln_kernel<<<BB * NN, 256, 0, stream>>>(tmp, nullptr, ln0_g, ln0_b, hbuf);

    dim3 gQKV(256 / 64, HDK / 64);     // 4 x 8
    dim3 gD(256 / 64, DD / 64);        // 4 x 12

    for (int l = 0; l < LL; ++l) {
        const float* WQ = wq + (size_t)l * DD * HDK;
        const float* WK = wk + (size_t)l * DD * HDK;
        const float* WV = wv + (size_t)l * DD * HDK;
        const float* BQ = bq + (size_t)l * HDK;
        const float* BK = bk + (size_t)l * HDK;
        const float* BV = bv + (size_t)l * HDK;
        const float* FCW = fcw + (size_t)l * HDK * DD;
        const float* FCB = fcb + (size_t)l * DD;
        const float* W1 = w1 + (size_t)l * DD * DINNER_;
        const float* B1 = b1 + (size_t)l * DINNER_;
        const float* W2 = w2 + (size_t)l * DINNER_ * DD;
        const float* B2 = b2 + (size_t)l * DD;

        gemm_bf16_wmma<0><<<gQKV, 256, 0, stream>>>(hbuf, WQ, BQ, qb, BB * NN, HDK, DD);
        gemm_bf16_wmma<0><<<gQKV, 256, 0, stream>>>(hbuf, WK, BK, kb, BB * NN, HDK, DD);
        gemm_bf16_wmma<0><<<gQKV, 256, 0, stream>>>(hbuf, WV, BV, vb, BB * NN, HDK, DD);

        attn_wmma<<<BB * HH, 128, 0, stream>>>(qb, kb, vb, smap, ob, out_att, (l == LL - 1) ? 1 : 0);

        gemm_bf16_wmma<0><<<gD, 256, 0, stream>>>(ob, FCW, FCB, tmp, BB * NN, DD, HDK);
        ln_kernel<<<BB * NN, 256, 0, stream>>>(tmp, hbuf, ln1_g + (size_t)l * DD, ln1_b + (size_t)l * DD, h2);

        gemm_bf16_wmma<1><<<gD, 256, 0, stream>>>(h2, W1, B1, ff1, BB * NN, DINNER_, DD);
        gemm_bf16_wmma<0><<<gD, 256, 0, stream>>>(ff1, W2, B2, tmp, BB * NN, DD, DINNER_);
        ln_kernel<<<BB * NN, 256, 0, stream>>>(tmp, h2, ln2_g + (size_t)l * DD, ln2_b + (size_t)l * DD, hbuf);
    }

    sp_kernel<<<BB * NN, 256, 0, stream>>>(hbuf, spw, spb, out_sp);
    type_kernel<<<BB * 3, 256, 0, stream>>>(qm, tw, tb, out_type);
}
